// TopKSAE_6184752906538
// MI455X (gfx1250) — matrix operand order repaired and checked by reading, not compiled, not run on hardware
//
#include <hip/hip_runtime.h>
#include <hip/hip_bf16.h>

typedef __attribute__((ext_vector_type(16))) __bf16 bf16x16;
typedef __attribute__((ext_vector_type(8)))  float  floatx8;
typedef __attribute__((ext_vector_type(4)))  unsigned int uint32x4;
typedef __attribute__((ext_vector_type(8)))  int int32x8;
typedef __attribute__((ext_vector_type(4)))  int int32x4;

#define INPUT_DIM 768
#define DICT_SIZE 32768
#define TOPK      32
#define BATCH     8192

#define RECON_ELEMS ((size_t)BATCH * INPUT_DIM)          // 6,291,456
#define ENC_ELEMS   ((size_t)BATCH * DICT_SIZE)          // 268,435,456

#define BM 128
#define BN 128
#define BK 32
#define FSTRIDE 34   // f32 units per LDS row (32 data + 2 TDM pad) -> even, conflict-free

// ---------------------------------------------------------------------------
// TDM: DMA one 128-row x 32-col f32 tile (row stride 768 f32) into LDS with
// 2-dword padding per 32-dword row (ISA ch.8 descriptor layout).
//   group0: count=1 | lds_addr | global_addr(57b) | type=2
//   group1: data_size=4B, pad_enable, pad_interval=32dw(code 4),
//           pad_amount=2dw(code 1), tensor_dim0=768, tile 32x128, stride 768
// 6-arg builtin variant (clang-23 / therock-10.0 HIP headers).
// ---------------------------------------------------------------------------
__device__ __forceinline__ void tdm_load_tile_f32(const float* gtile,
                                                  unsigned lds_byte_off,
                                                  unsigned tensor_rows)
{
    unsigned long long ga = (unsigned long long)(const void*)gtile;
    uint32x4 g0;
    g0[0] = 1u;                                        // count=1, user desc
    g0[1] = lds_byte_off;                              // lds_addr [63:32]
    g0[2] = (unsigned)(ga & 0xFFFFFFFFu);              // global_addr lo
    g0[3] = (unsigned)((ga >> 32) & 0x01FFFFFFu)       // global_addr [56:32]
          | (2u << 30);                                // type=2 ("image")
    int32x8 g1;
    g1[0] = (int)((2u << 16)        // data_size = 4 bytes
                | (1u << 20)        // pad_enable
                | (4u << 22)        // pad_interval: every 32 dwords
                | (1u << 25));      // pad_amount: 2 dwords
    g1[1] = (int)((768u & 0xFFFFu) << 16);             // tensor_dim0 lo16
    g1[2] = (int)(((768u >> 16) & 0xFFFFu)             // tensor_dim0 hi16
                | ((tensor_rows & 0xFFFFu) << 16));    // tensor_dim1 lo16
    g1[3] = (int)(((tensor_rows >> 16) & 0xFFFFu)      // tensor_dim1 hi16
                | ((unsigned)BK << 16));               // tile_dim0 = 32
    g1[4] = BM;                                        // tile_dim1 = 128
    g1[5] = 768;                                       // tensor_dim0_stride
    g1[6] = 0;
    g1[7] = 0;
    int32x4 z4 = {0, 0, 0, 0};                         // 2-D: groups 2/3 unused
    int32x8 z8 = {0, 0, 0, 0, 0, 0, 0, 0};
    __builtin_amdgcn_tensor_load_to_lds(g0, g1, z4, z4, z8, 0);
}

// ---------------------------------------------------------------------------
// Kernel 1: z = X @ W_enc^T + b_enc  via bf16 WMMA (f32 accumulate).
// 128x128 tile per 128-thread block; 4 waves in 2x2 quadrants, each wave a
// 4x4 grid of v_wmma_f32_16x16x32_bf16 accumulators over K in chunks of 32.
// A/B tiles arrive by Tensor Data Mover (tensor_load_to_lds, TENSORcnt);
// f32 -> bf16 conversion is fused into fragment construction.
// ---------------------------------------------------------------------------
__global__ __launch_bounds__(128)
void sae_encode_wmma(const float* __restrict__ X,
                     const float* __restrict__ Wenc,
                     const float* __restrict__ benc,
                     float* __restrict__ Z)
{
    __shared__ float sAf[BM * FSTRIDE];   // 128 x 34 f32 (TDM-padded rows)
    __shared__ float sBf[BN * FSTRIDE];

    const int tid  = threadIdx.x;
    const int lane = tid & 31;
    const int wave = tid >> 5;
    const int m0   = blockIdx.y * BM;
    const int n0   = blockIdx.x * BN;
    const int wm   = (wave >> 1) * 64;     // wave quadrant
    const int wn   = (wave & 1) * 64;
    const int hh   = lane >> 4;            // half-wave selector (ISA layout)
    const int l15  = lane & 15;

    const unsigned ldsA = (unsigned)(unsigned long long)(uintptr_t)(void*)&sAf[0];
    const unsigned ldsB = (unsigned)(unsigned long long)(uintptr_t)(void*)&sBf[0];

    floatx8 acc[4][4] = {};

    for (int kk = 0; kk < INPUT_DIM; kk += BK) {
        // ---- TDM fetch of both tiles, issued by wave 0 only ----
        if (wave == 0) {
            tdm_load_tile_f32(&X[(long)m0 * INPUT_DIM + kk],    ldsA, BATCH);
            tdm_load_tile_f32(&Wenc[(long)n0 * INPUT_DIM + kk], ldsB, DICT_SIZE);
            __builtin_amdgcn_s_wait_tensorcnt(0);
        }
        __syncthreads();   // publish tiles to all 4 waves

        // B fragments (32x16 bf16): lanes = N column; lanes 0-15 K=0..15,
        // lanes 16-31 K=16..31, VGPR i -> K = 2i (+16 for upper half).
        bf16x16 bfrag[4];
        #pragma unroll
        for (int ni = 0; ni < 4; ++ni) {
            const float* bp = &sBf[(wn + ni * 16 + l15) * FSTRIDE + 16 * hh];
            #pragma unroll
            for (int i = 0; i < 8; ++i) {
                float2 p = *reinterpret_cast<const float2*>(bp + 2 * i);
                bfrag[ni][2 * i]     = (__bf16)p.x;
                bfrag[ni][2 * i + 1] = (__bf16)p.y;
            }
        }

        // Stream A fragments (16x32 bf16): lane<16 row M=l15 holds
        // K {0..15}; lane>=16 holds K {16..31}.
        #pragma unroll
        for (int mi = 0; mi < 4; ++mi) {
            const float* ap = &sAf[(wm + mi * 16 + l15) * FSTRIDE + 16 * hh];
            bf16x16 afrag;
            #pragma unroll
            for (int i = 0; i < 8; ++i) {
                int k = 2 * i;   // {0,2,4,6,8,10,12,14}
                float2 p = *reinterpret_cast<const float2*>(ap + k);
                afrag[2 * i]     = (__bf16)p.x;
                afrag[2 * i + 1] = (__bf16)p.y;
            }
            #pragma unroll
            for (int ni = 0; ni < 4; ++ni)
                acc[mi][ni] = __builtin_amdgcn_wmma_f32_16x16x32_bf16(
                    false, afrag, false, bfrag[ni],
                    (short)0, acc[mi][ni], false, false);
        }

        __syncthreads();   // all reads done before next TDM overwrites tiles
    }

    // D layout: lane l -> n = l&15; VGPR i -> m = i + 8*(l>>4)
    #pragma unroll
    for (int ni = 0; ni < 4; ++ni) {
        int gn = n0 + wn + ni * 16 + l15;
        float bias = benc[gn];
        #pragma unroll
        for (int mi = 0; mi < 4; ++mi) {
            #pragma unroll
            for (int i = 0; i < 8; ++i) {
                int gm = m0 + wm + mi * 16 + i + 8 * hh;
                Z[(long)gm * DICT_SIZE + gn] = acc[mi][ni][i] + bias;
            }
        }
    }
}

// ---------------------------------------------------------------------------
// Kernel 2: per-row TopK (iterative argmax, 32 passes over L2-resident row),
// sparse-scatter the encoded output, gathered decode, per-row squared error.
// One 256-thread block per row.  Scans vectorized to float4.
// ---------------------------------------------------------------------------
__global__ __launch_bounds__(256)
void sae_topk_decode(const float* __restrict__ X,
                     const float* __restrict__ Wdec,
                     const float* __restrict__ bdec,
                     float* __restrict__ Z,          // encoded region (in/out)
                     float* __restrict__ recon,
                     float* __restrict__ row_loss)
{
    __shared__ float rv[256];
    __shared__ int   ri[256];
    __shared__ float s_val[TOPK];
    __shared__ int   s_idx[TOPK];

    const int row = blockIdx.x;
    const int tid = threadIdx.x;
    float* zrow = Z + (long)row * DICT_SIZE;
    float4* zrow4 = reinterpret_cast<float4*>(zrow);

    for (int sel = 0; sel < TOPK; ++sel) {
        float bv = -__builtin_inff();
        int   bi = 0;
        for (int j4 = tid; j4 < DICT_SIZE / 4; j4 += 256) {
            float4 v = zrow4[j4];
            int jb = j4 * 4;
            if (v.x > bv) { bv = v.x; bi = jb;     }
            if (v.y > bv) { bv = v.y; bi = jb + 1; }
            if (v.z > bv) { bv = v.z; bi = jb + 2; }
            if (v.w > bv) { bv = v.w; bi = jb + 3; }
        }
        rv[tid] = bv; ri[tid] = bi;
        __syncthreads();
        for (int off = 128; off > 0; off >>= 1) {
            if (tid < off) {
                float ov = rv[tid + off]; int oi = ri[tid + off];
                if (ov > rv[tid] || (ov == rv[tid] && oi < ri[tid])) {
                    rv[tid] = ov; ri[tid] = oi;
                }
            }
            __syncthreads();
        }
        if (tid == 0) {
            s_val[sel] = rv[0];
            s_idx[sel] = ri[0];
            zrow[ri[0]] = -__builtin_inff();   // exclude from next pass
            __threadfence();
        }
        __syncthreads();
    }

    // encoded = zeros, then scatter the K kept (value, index) pairs
    const float4 z4 = make_float4(0.f, 0.f, 0.f, 0.f);
    for (int j4 = tid; j4 < DICT_SIZE / 4; j4 += 256) zrow4[j4] = z4;
    __syncthreads();
    if (tid < TOPK) zrow[s_idx[tid]] = s_val[tid];

    // decode: recon[d] = b_dec[d] + sum_k val[k] * W_dec[d, idx[k]]
    float lsum = 0.0f;
    for (int d = tid; d < INPUT_DIM; d += 256) {
        float a = bdec[d];
        const float* wrow = Wdec + (long)d * DICT_SIZE;
        #pragma unroll
        for (int k = 0; k < TOPK; ++k) a += s_val[k] * wrow[s_idx[k]];
        recon[(long)row * INPUT_DIM + d] = a;
        float diff = a - X[(long)row * INPUT_DIM + d];
        lsum += diff * diff;
    }
    rv[tid] = lsum;
    __syncthreads();
    for (int off = 128; off > 0; off >>= 1) {
        if (tid < off) rv[tid] += rv[tid + off];
        __syncthreads();
    }
    if (tid == 0) row_loss[row] = rv[0];
}

// ---------------------------------------------------------------------------
// Kernel 3: deterministic reduction of per-row losses -> mean.
// ---------------------------------------------------------------------------
__global__ __launch_bounds__(256)
void sae_loss_reduce(const float* __restrict__ row_loss,
                     float* __restrict__ out_loss)
{
    __shared__ float rv[256];
    float s = 0.0f;
    for (int i = threadIdx.x; i < BATCH; i += 256) s += row_loss[i];
    rv[threadIdx.x] = s;
    __syncthreads();
    for (int off = 128; off > 0; off >>= 1) {
        if (threadIdx.x < off) rv[threadIdx.x] += rv[threadIdx.x + off];
        __syncthreads();
    }
    if (threadIdx.x == 0) out_loss[0] = rv[0] / (float)BATCH;
}

// ---------------------------------------------------------------------------
extern "C" void kernel_launch(void* const* d_in, const int* in_sizes, int n_in,
                              void* d_out, int out_size, void* d_ws, size_t ws_size,
                              hipStream_t stream)
{
    const float* X    = (const float*)d_in[0];
    const float* Wenc = (const float*)d_in[1];
    const float* benc = (const float*)d_in[2];
    const float* Wdec = (const float*)d_in[3];
    const float* bdec = (const float*)d_in[4];

    float* out   = (float*)d_out;
    float* recon = out;                            // (8192, 768)
    float* enc   = out + RECON_ELEMS;              // (8192, 32768)
    float* loss  = out + RECON_ELEMS + ENC_ELEMS;  // scalar
    float* row_loss = (float*)d_ws;                // 8192 floats of scratch

    dim3 g1(DICT_SIZE / BN, BATCH / BM);           // (256, 64) tiles
    sae_encode_wmma<<<g1, 128, 0, stream>>>(X, Wenc, benc, enc);
    sae_topk_decode<<<BATCH, 256, 0, stream>>>(X, Wdec, bdec, enc, recon, row_loss);
    sae_loss_reduce<<<1, 256, 0, stream>>>(row_loss, loss);
}
